// RepRNN_51694226375301
// MI455X (gfx1250) — compile-verified, matrix operand order checked by reading, not run on hardware
//
#include <hip/hip_runtime.h>
#include <cstdint>
#include <cstddef>

// ---------------------------------------------------------------------------
// RepRNN for MI455X (gfx1250): bf16 WMMA batched Gram + persistent WMMA GRU
// ---------------------------------------------------------------------------

typedef __attribute__((ext_vector_type(16))) __bf16 v16bf;
typedef __attribute__((ext_vector_type(8)))  float  v8f;

union Frag16 {
    v16bf v;
    uint4 q[2];
};

#define S_LEN 512
#define BATCH 128
#define DIM   300
#define DPAD  320
#define HID   512

// ---------------------------------------------------------------------------
// Phase A: gather embed rows by `ends`, L2-normalize over D, store bf16
// layout xn[b][s][DPAD] (zero-padded 300..319).  One wave per (s,b).
// ---------------------------------------------------------------------------
__global__ void gather_norm_kernel(const int* __restrict__ ends,
                                   const float* __restrict__ embed,
                                   __bf16* __restrict__ xn) {
    const int blk  = blockIdx.x;          // s*BATCH + b
    const int s    = blk >> 7;
    const int b    = blk & 127;
    const int lane = threadIdx.x;
    const int row  = ends[blk];
    const float* e = embed + (size_t)row * DIM;

    float ss = 0.f;
    for (int d = lane; d < DIM; d += 32) { float v = e[d]; ss += v * v; }
    #pragma unroll
    for (int m = 1; m < 32; m <<= 1) ss += __shfl_xor(ss, m, 32);
    const float scale = 1.0f / fmaxf(sqrtf(ss), 1e-8f);

    __bf16* out = xn + ((size_t)b * S_LEN + s) * DPAD;
    for (int d = lane; d < DPAD; d += 32) {
        float v = (d < DIM) ? e[d] * scale : 0.f;
        out[d] = (__bf16)v;
    }
}

// ---------------------------------------------------------------------------
// Phase B: per batch, causal row-max of the 512x512 Gram matrix of xn.
// One wave per (batch, 16-row i-tile).  A-fragments preloaded (10 k-steps of
// K=32 covering DPAD=320); loop over j-tiles <= i-tile with one
// v_wmma_f32_16x16x32_bf16 per k-step, masked max, then cross-lane max.
// ---------------------------------------------------------------------------
__global__ void simmax_kernel(const __bf16* __restrict__ xn,
                              float* __restrict__ feats) {
    const int b    = blockIdx.x >> 5;
    const int it   = blockIdx.x & 31;
    const int lane = threadIdx.x;
    const int half = lane >> 4;
    const int l15  = lane & 15;
    const int koff = half ? 8 : 0;
    const __bf16* Xb = xn + (size_t)b * S_LEN * DPAD;
    const int i0 = it * 16;

    Frag16 A[10];
    {
        const __bf16* ap = Xb + (size_t)(i0 + l15) * DPAD + koff;
        #pragma unroll
        for (int ks = 0; ks < 10; ++ks) {
            A[ks].q[0] = *(const uint4*)(ap + ks * 32);
            A[ks].q[1] = *(const uint4*)(ap + ks * 32 + 16);
        }
    }

    float rmax[8];
    #pragma unroll
    for (int v = 0; v < 8; ++v) rmax[v] = -3.4e38f;

    for (int jt = 0; jt <= it; ++jt) {
        v8f acc = {};
        const __bf16* bp = Xb + (size_t)(jt * 16 + l15) * DPAD + koff;
        #pragma unroll
        for (int ks = 0; ks < 10; ++ks) {
            Frag16 Bf;
            Bf.q[0] = *(const uint4*)(bp + ks * 32);
            Bf.q[1] = *(const uint4*)(bp + ks * 32 + 16);
            acc = __builtin_amdgcn_wmma_f32_16x16x32_bf16(
                false, A[ks].v, false, Bf.v, (short)0, acc, false, false);
        }
        const int j = jt * 16 + l15;
        #pragma unroll
        for (int v = 0; v < 8; ++v) {
            const int i = i0 + v + (half ? 8 : 0);
            if (j < i) rmax[v] = fmaxf(rmax[v], acc[v]);
        }
    }

    #pragma unroll
    for (int v = 0; v < 8; ++v) {
        float m = rmax[v];
        #pragma unroll
        for (int d = 1; d < 16; d <<= 1) m = fmaxf(m, __shfl_xor(m, d, 32));
        if (l15 == 0) {
            const int i = i0 + v + (half ? 8 : 0);
            feats[i * BATCH + b] = (i == 0) ? 0.0f : m;
        }
    }
}

// ---------------------------------------------------------------------------
// Device-wide sense-reversing barrier (persistent GRU kernel, 32 blocks).
// ---------------------------------------------------------------------------
__device__ __forceinline__ void grid_sync(volatile int* cnt, volatile int* gen,
                                          int nb) {
    __syncthreads();
    if (threadIdx.x == 0) {
        __threadfence();
        const int g = *gen;
        if (atomicAdd((int*)cnt, 1) == nb - 1) {
            *cnt = 0;
            __threadfence();
            atomicAdd((int*)gen, 1);
        } else {
            while (*gen == g) { __builtin_amdgcn_s_sleep(1); }
        }
        __threadfence();
    }
    __syncthreads();
}

// ---------------------------------------------------------------------------
// Phase C: persistent GRU.  32 blocks x 256 threads (8 waves).  Block owns a
// 16-wide H-column slice; wave w owns batch rows [16w,16w+16).  w_hh^T is
// pre-swizzled once into WMMA B-fragment layout in LDS (48 KB).  h kept in
// f32 registers (matches the C/D fragment layout); bf16 copy ping-pongs in
// global for other blocks' A-fragments.  One grid barrier per step.
// ---------------------------------------------------------------------------
__global__ void __launch_bounds__(256, 1)
gru_kernel(const float* __restrict__ feats,
           const float* __restrict__ w_ih, const float* __restrict__ w_hh,
           const float* __restrict__ b_ih, const float* __restrict__ b_hh,
           __bf16* __restrict__ hb0, __bf16* __restrict__ hb1,
           float* __restrict__ hfinal, int* __restrict__ syncp) {
    __shared__ __bf16 ldsB[3 * 16 * 32 * 16];   // 48 KB: [gate][kstep][lane][16]

    const int c0   = blockIdx.x * 16;
    const int tid  = threadIdx.x;
    const int wave = tid >> 5;
    const int lane = tid & 31;
    const int half = lane >> 4;
    const int l15  = lane & 15;
    const int koff = half ? 8 : 0;

    // Stage w_hh^T into fragment-linear LDS (step-invariant).
    for (int f = tid; f < 3 * 16 * 32; f += 256) {
        const int g    = f / (16 * 32);
        const int ks   = (f >> 5) & 15;
        const int ln   = f & 31;
        const int gcol = g * HID + c0 + (ln & 15);
        const int ko   = (ln >> 4) ? 8 : 0;
        __bf16* dst = ldsB + f * 16;
        #pragma unroll
        for (int e = 0; e < 16; ++e) {
            const int k = ks * 32 + ko + ((e < 8) ? e : (e + 8));
            dst[e] = (__bf16)w_hh[(size_t)gcol * HID + k];
        }
    }
    // Zero-init h bf16 buffer 0 for our column slice (all 128 rows).
    for (int t = tid; t < BATCH * 16; t += 256) {
        hb0[(size_t)(t >> 4) * HID + c0 + (t & 15)] = (__bf16)0.0f;
    }

    // Per-lane gate constants (our column is fixed per lane).
    const int hcol = c0 + l15;
    const float wih_r = w_ih[0 * HID + hcol], wih_z = w_ih[1 * HID + hcol],
                wih_n = w_ih[2 * HID + hcol];
    const float bih_r = b_ih[0 * HID + hcol], bih_z = b_ih[1 * HID + hcol],
                bih_n = b_ih[2 * HID + hcol];
    const float bhh_r = b_hh[0 * HID + hcol], bhh_z = b_hh[1 * HID + hcol],
                bhh_n = b_hh[2 * HID + hcol];

    float hreg[8];
    #pragma unroll
    for (int v = 0; v < 8; ++v) hreg[v] = 0.0f;

    const int arow = wave * 16 + l15;

    __syncthreads();
    grid_sync(syncp, syncp + 1, gridDim.x);

    for (int s = 0; s < S_LEN; ++s) {
        const __bf16* hin  = (s & 1) ? hb1 : hb0;
        __bf16*       hout = (s & 1) ? hb0 : hb1;

        v8f ar = {}, az = {}, an = {};
        const __bf16* ap = hin + (size_t)arow * HID + koff;
        #pragma unroll
        for (int ks = 0; ks < 16; ++ks) {
            Frag16 a;
            a.q[0] = *(const uint4*)(ap + ks * 32);
            a.q[1] = *(const uint4*)(ap + ks * 32 + 16);
            Frag16 br, bz, bn;
            const __bf16* lr = ldsB + (((0 * 16 + ks) * 32 + lane) << 4);
            const __bf16* lz = ldsB + (((1 * 16 + ks) * 32 + lane) << 4);
            const __bf16* ln_ = ldsB + (((2 * 16 + ks) * 32 + lane) << 4);
            br.q[0] = *(const uint4*)(lr);  br.q[1] = *(const uint4*)(lr + 8);
            bz.q[0] = *(const uint4*)(lz);  bz.q[1] = *(const uint4*)(lz + 8);
            bn.q[0] = *(const uint4*)(ln_); bn.q[1] = *(const uint4*)(ln_ + 8);
            ar = __builtin_amdgcn_wmma_f32_16x16x32_bf16(
                false, a.v, false, br.v, (short)0, ar, false, false);
            az = __builtin_amdgcn_wmma_f32_16x16x32_bf16(
                false, a.v, false, bz.v, (short)0, az, false, false);
            an = __builtin_amdgcn_wmma_f32_16x16x32_bf16(
                false, a.v, false, bn.v, (short)0, an, false, false);
        }

        #pragma unroll
        for (int v = 0; v < 8; ++v) {
            const int brow = wave * 16 + v + (half ? 8 : 0);
            const float x  = feats[s * BATCH + brow];
            const float r  = 1.0f / (1.0f + expf(-(x * wih_r + bih_r + ar[v] + bhh_r)));
            const float z  = 1.0f / (1.0f + expf(-(x * wih_z + bih_z + az[v] + bhh_z)));
            const float hn = an[v] + bhh_n;
            const float nn = tanhf(x * wih_n + bih_n + r * hn);
            const float hnew = (1.0f - z) * nn + z * hreg[v];
            hreg[v] = hnew;
            hout[(size_t)brow * HID + hcol] = (__bf16)hnew;
        }
        grid_sync(syncp, syncp + 1, gridDim.x);
    }

    #pragma unroll
    for (int v = 0; v < 8; ++v) {
        const int brow = wave * 16 + v + (half ? 8 : 0);
        hfinal[(size_t)brow * HID + hcol] = hreg[v];
    }
}

// ---------------------------------------------------------------------------
// Phase D: out[b] = dot(h_last[b], fc_w) + fc_b.  One wave per batch row.
// ---------------------------------------------------------------------------
__global__ void fc_kernel(const float* __restrict__ hfin,
                          const float* __restrict__ fc_w,
                          const float* __restrict__ fc_b,
                          float* __restrict__ out) {
    const int b    = blockIdx.x;
    const int lane = threadIdx.x;
    float s = 0.f;
    for (int k = lane; k < HID; k += 32) s += hfin[(size_t)b * HID + k] * fc_w[k];
    #pragma unroll
    for (int m = 1; m < 32; m <<= 1) s += __shfl_xor(s, m, 32);
    if (lane == 0) out[b] = s + fc_b[0];
}

// ---------------------------------------------------------------------------
extern "C" void kernel_launch(void* const* d_in, const int* in_sizes, int n_in,
                              void* d_out, int out_size, void* d_ws, size_t ws_size,
                              hipStream_t stream) {
    (void)in_sizes; (void)n_in; (void)out_size; (void)ws_size;
    const int*   ends  = (const int*)  d_in[1];
    const float* embed = (const float*)d_in[3];
    const float* w_ih  = (const float*)d_in[4];
    const float* w_hh  = (const float*)d_in[5];
    const float* b_ih  = (const float*)d_in[6];
    const float* b_hh  = (const float*)d_in[7];
    const float* fc_w  = (const float*)d_in[8];
    const float* fc_b  = (const float*)d_in[9];
    float* out = (float*)d_out;

    char* ws = (char*)d_ws;
    size_t off = 0;
    auto alloc = [&](size_t bytes) -> void* {
        void* p = ws + off;
        off = (off + bytes + 255) & ~(size_t)255;
        return p;
    };
    __bf16* xn    = (__bf16*)alloc((size_t)BATCH * S_LEN * DPAD * 2); // 40 MB
    float*  feats = (float*) alloc((size_t)S_LEN * BATCH * 4);
    __bf16* hb0   = (__bf16*)alloc((size_t)BATCH * HID * 2);
    __bf16* hb1   = (__bf16*)alloc((size_t)BATCH * HID * 2);
    float*  hfin  = (float*) alloc((size_t)BATCH * HID * 4);
    int*    syncp = (int*)   alloc(256);

    hipMemsetAsync(syncp, 0, 2 * sizeof(int), stream);
    gather_norm_kernel<<<S_LEN * BATCH, 32, 0, stream>>>(ends, embed, xn);
    simmax_kernel<<<BATCH * 32, 32, 0, stream>>>(xn, feats);
    gru_kernel<<<32, 256, 0, stream>>>(feats, w_ih, w_hh, b_ih, b_hh,
                                       hb0, hb1, hfin, syncp);
    fc_kernel<<<BATCH, 32, 0, stream>>>(hfin, fc_w, fc_b, out);
}
